// SynthesisModel_15487652070022
// MI455X (gfx1250) — matrix operand order repaired and checked by reading, not compiled, not run on hardware
//
#include <hip/hip_runtime.h>

typedef __attribute__((ext_vector_type(16))) __bf16 v16bf;
typedef __attribute__((ext_vector_type(8)))  float  v8f;

#define B_     32
#define T_     512
#define U_     64
#define V_     60
#define H_     400
#define K_     10
#define M_     20
#define G3_    1200   // 3*H
#define HP_    416    // H padded to 13*32
#define IN0P_  64     // layer0 input (3+60=63) padded
#define INLP_  480    // layer1/2 input (3+400+60=463) padded to 15*32
#define FEATP_ 1216   // 3*H padded to 38*32
#define NHEAD_ 128    // 121 head outputs padded

#define WMMA(acc, a, b) \
  (acc) = __builtin_amdgcn_wmma_f32_16x16x32_bf16(false, (a), false, (b), (short)0, (acc), false, false)

static __device__ __forceinline__ float sigmoidf_(float v) { return 1.0f / (1.0f + __expf(-v)); }

// A fragment: row-major MxK, element (m,k) at A[m*lda+k].  ISA 7.12.2 16-bit A 16x32 layout.
static __device__ __forceinline__ v16bf ldA(const __bf16* A, int lda, int m0, int k0) {
  int lane = threadIdx.x & 31;
  int m = m0 + (lane & 15);
  int g = lane >> 4;
  v16bf r;
#pragma unroll
  for (int j = 0; j < 8; ++j) {
    int k = k0 + ((j >> 2) << 4) + ((j & 3) << 1) + (g << 3);
    r[2 * j]     = A[m * lda + k];
    r[2 * j + 1] = A[m * lda + k + 1];
  }
  return r;
}

// B fragment with B(k,n) = W[n*ldk + k]  (W stored row-major [N][ldk]; GEMM uses W^T).
static __device__ __forceinline__ v16bf ldB(const __bf16* W, int ldk, int n0, int k0) {
  int lane = threadIdx.x & 31;
  int n = n0 + (lane & 15);
  int g = lane >> 4;
  v16bf r;
#pragma unroll
  for (int j = 0; j < 8; ++j) {
    int k = k0 + ((j >> 2) << 4) + ((j & 3) << 1) + (g << 3);
    r[2 * j]     = W[n * ldk + k];
    r[2 * j + 1] = W[n * ldk + k + 1];
  }
  return r;
}

// Layer-0 A fragment built on the fly from x_t (global f32) and window w (LDS f32).
static __device__ __forceinline__ v16bf ldAi0(const float* x, int t, const float* w, int m0, int k0) {
  int lane = threadIdx.x & 31;
  int m = m0 + (lane & 15);
  int g = lane >> 4;
  v16bf r;
#pragma unroll
  for (int j = 0; j < 8; ++j) {
    int k = k0 + ((j >> 2) << 4) + ((j & 3) << 1) + (g << 3);
#pragma unroll
    for (int h = 0; h < 2; ++h) {
      int kk = k + h;
      float v = 0.0f;
      if (kk < 3)       v = x[((size_t)m * T_ + t) * 3 + kk];
      else if (kk < 63) v = w[m * V_ + (kk - 3)];
      r[2 * j + h] = (__bf16)v;
    }
  }
  return r;
}

// ---------------- weight / activation packing ----------------

extern "C" __global__ void pack_weight_bf16(__bf16* dst, const float* src,
                                            int rs, int cs, int rd, int cd) {
  int i = blockIdx.x * blockDim.x + threadIdx.x;
  if (i >= rd * cd) return;
  int r = i / cd, c = i % cd;
  float v = (r < rs && c < cs) ? src[r * cs + c] : 0.0f;
  dst[i] = (__bf16)v;
}

// Ain layout [t][b][INLP_] = concat(x_t, out_prev_t, w_t) padded, bf16.
extern "C" __global__ void pack_ain(__bf16* dst, const float* x,
                                    const float* outprev, const float* wseq) {
  size_t i = (size_t)blockIdx.x * blockDim.x + threadIdx.x;
  if (i >= (size_t)T_ * B_ * INLP_) return;
  int k  = (int)(i % INLP_);
  int bt = (int)(i / INLP_);
  int b  = bt % B_;
  int t  = bt / B_;
  float v = 0.0f;
  if (k < 3)        v = x[((size_t)b * T_ + t) * 3 + k];
  else if (k < 403) v = outprev[((size_t)b * T_ + t) * H_ + (k - 3)];
  else if (k < 463) v = wseq[((size_t)b * T_ + t) * V_ + (k - 403)];
  dst[i] = (__bf16)v;
}

// feat layout [b*T+t][FEATP_] = concat(out0,out1,out2) padded, bf16.
extern "C" __global__ void pack_feat(__bf16* dst, const float* o0,
                                     const float* o1, const float* o2) {
  size_t i = (size_t)blockIdx.x * blockDim.x + threadIdx.x;
  if (i >= (size_t)B_ * T_ * FEATP_) return;
  int k = (int)(i % FEATP_);
  size_t r = i / FEATP_;           // r = b*T + t
  float v = 0.0f;
  if (k < 400)       v = o0[r * H_ + k];
  else if (k < 800)  v = o1[r * H_ + (k - 400)];
  else if (k < 1200) v = o2[r * H_ + (k - 800)];
  dst[i] = (__bf16)v;
}

// ---------------- layer 0: attention GRU, persistent workgroup ----------------

extern "C" __global__ __launch_bounds__(1024, 1) void layer0_seq(
    const float* __restrict__ x, const int* __restrict__ c,
    const __bf16* __restrict__ Wih, const __bf16* __restrict__ Whh,
    const float* __restrict__ bih, const float* __restrict__ bhh,
    const float* __restrict__ Ww, const float* __restrict__ bw,
    float* __restrict__ out0, float* __restrict__ wseq, float* __restrict__ phit)
{
  __shared__ __bf16 hbf[2][B_][HP_];   // double-buffered hidden state (bf16)
  __shared__ float  wls[B_][V_];       // attention window vector
  __shared__ float  kap[B_][K_];
  __shared__ float  alp[B_][K_];
  __shared__ float  bet[B_][K_];
  int tid = threadIdx.x, wave = tid >> 5, lane = tid & 31;

  for (int i = tid; i < 2 * B_ * HP_; i += 1024) ((__bf16*)hbf)[i] = (__bf16)0.0f;
  for (int i = tid; i < B_ * V_; i += 1024) ((float*)wls)[i] = 0.0f;
  for (int i = tid; i < B_ * K_; i += 1024) ((float*)kap)[i] = 0.0f;
  __syncthreads();

  int cur = 0;
  for (int t = 0; t < T_; ++t) {
    const __bf16* hc = &hbf[cur][0][0];
    // ---- GRU gates: 2 m-tiles x 25 column groups, one task per wave round ----
    for (int task = wave; task < 50; task += 32) {
      int mt = task / 25, jg = task % 25;
      int cr = jg * 16, cz = H_ + jg * 16, cn = 2 * H_ + jg * 16;
      v8f ar = {}, az = {}, ai = {}, ah = {};
      for (int kb = 0; kb < IN0P_ / 32; ++kb) {          // input projection
        v16bf a = ldAi0(x, t, &wls[0][0], mt * 16, kb * 32);
        WMMA(ar, a, ldB(Wih, IN0P_, cr, kb * 32));
        WMMA(az, a, ldB(Wih, IN0P_, cz, kb * 32));
        WMMA(ai, a, ldB(Wih, IN0P_, cn, kb * 32));
      }
      for (int kb = 0; kb < HP_ / 32; ++kb) {            // recurrent projection
        v16bf a = ldA(hc, HP_, mt * 16, kb * 32);
        WMMA(ar, a, ldB(Whh, HP_, cr, kb * 32));
        WMMA(az, a, ldB(Whh, HP_, cz, kb * 32));
        WMMA(ah, a, ldB(Whh, HP_, cn, kb * 32));
      }
      int g = lane >> 4;
      int col = jg * 16 + (lane & 15);
      float br = bih[col] + bhh[col];
      float bz = bih[H_ + col] + bhh[H_ + col];
      float bi = bih[2 * H_ + col];
      float bn = bhh[2 * H_ + col];
#pragma unroll
      for (int j = 0; j < 8; ++j) {
        int b = mt * 16 + j + 8 * g;
        float r = sigmoidf_(ar[j] + br);
        float z = sigmoidf_(az[j] + bz);
        float n = tanhf(ai[j] + bi + r * (ah[j] + bn));
        float ho = (float)hbf[cur][b][col];
        float hn = (1.0f - z) * n + z * ho;
        out0[((size_t)b * T_ + t) * H_ + col] = hn;
        hbf[cur ^ 1][b][col] = (__bf16)hn;
      }
    }
    __syncthreads();
    cur ^= 1;
    // ---- attention mixture params: p = exp(h @ Ww.T + bw) ----
    if (tid < B_ * 30) {
      int b = tid / 30, q = tid % 30;
      float acc = bw[q];
      const __bf16* hb = &hbf[cur][b][0];
      for (int j = 0; j < H_; ++j) acc += (float)hb[j] * Ww[q * H_ + j];
      float p = __expf(acc);
      if (q < K_)          alp[b][q] = p;
      else if (q < 2 * K_) bet[b][q - K_] = p;
      else                 kap[b][q - 2 * K_] += p;   // kappa += dk
    }
    __syncthreads();
    // ---- window vector (deterministic gather over one-hot c) ----
    for (int i = tid; i < B_ * V_; i += 1024) {
      int b = i / V_, v = i % V_;
      float acc = 0.0f;
      for (int u = 0; u < U_; ++u) {
        if (c[b * U_ + u] == v) {
          float phi = 0.0f;
          for (int q = 0; q < K_; ++q) {
            float d = kap[b][q] - (float)u;
            phi += alp[b][q] * __expf(-bet[b][q] * d * d);
          }
          acc += phi;
        }
      }
      wls[b][v] = acc;
      wseq[((size_t)b * T_ + t) * V_ + v] = acc;
    }
    if (t == T_ - 1) {
      for (int i = tid; i < B_ * 65; i += 1024) {
        int b = i / 65, u = i % 65;
        float phi = 0.0f;
        for (int q = 0; q < K_; ++q) {
          float d = kap[b][q] - (float)u;
          phi += alp[b][q] * __expf(-bet[b][q] * d * d);
        }
        phit[i] = phi;
      }
    }
    __syncthreads();
  }
}

// ---------------- layers 1/2: GRU with precomputed packed inputs ----------------

extern "C" __global__ __launch_bounds__(1024, 1) void layer_seq(
    const __bf16* __restrict__ Ain, const __bf16* __restrict__ Wih,
    const __bf16* __restrict__ Whh, const float* __restrict__ bih,
    const float* __restrict__ bhh, float* __restrict__ outl)
{
  __shared__ __bf16 hbf[2][B_][HP_];
  int tid = threadIdx.x, wave = tid >> 5, lane = tid & 31;
  for (int i = tid; i < 2 * B_ * HP_; i += 1024) ((__bf16*)hbf)[i] = (__bf16)0.0f;
  __syncthreads();
  int cur = 0;
  for (int t = 0; t < T_; ++t) {
    const __bf16* hc = &hbf[cur][0][0];
    const __bf16* At = Ain + (size_t)t * B_ * INLP_;
    for (int task = wave; task < 50; task += 32) {
      int mt = task / 25, jg = task % 25;
      int cr = jg * 16, cz = H_ + jg * 16, cn = 2 * H_ + jg * 16;
      v8f ar = {}, az = {}, ai = {}, ah = {};
      for (int kb = 0; kb < INLP_ / 32; ++kb) {
        v16bf a = ldA(At, INLP_, mt * 16, kb * 32);
        WMMA(ar, a, ldB(Wih, INLP_, cr, kb * 32));
        WMMA(az, a, ldB(Wih, INLP_, cz, kb * 32));
        WMMA(ai, a, ldB(Wih, INLP_, cn, kb * 32));
      }
      for (int kb = 0; kb < HP_ / 32; ++kb) {
        v16bf a = ldA(hc, HP_, mt * 16, kb * 32);
        WMMA(ar, a, ldB(Whh, HP_, cr, kb * 32));
        WMMA(az, a, ldB(Whh, HP_, cz, kb * 32));
        WMMA(ah, a, ldB(Whh, HP_, cn, kb * 32));
      }
      int g = lane >> 4;
      int col = jg * 16 + (lane & 15);
      float br = bih[col] + bhh[col];
      float bz = bih[H_ + col] + bhh[H_ + col];
      float bi = bih[2 * H_ + col];
      float bn = bhh[2 * H_ + col];
#pragma unroll
      for (int j = 0; j < 8; ++j) {
        int b = mt * 16 + j + 8 * g;
        float r = sigmoidf_(ar[j] + br);
        float z = sigmoidf_(az[j] + bz);
        float n = tanhf(ai[j] + bi + r * (ah[j] + bn));
        float ho = (float)hbf[cur][b][col];
        float hn = (1.0f - z) * n + z * ho;
        outl[((size_t)b * T_ + t) * H_ + col] = hn;
        hbf[cur ^ 1][b][col] = (__bf16)hn;
      }
    }
    __syncthreads();
    cur ^= 1;
  }
}

// ---------------- mixture-density head GEMM: (B*T x 1216) x (1216 x 128) ----------------

extern "C" __global__ void head_gemm(const __bf16* __restrict__ feat,
                                     const __bf16* __restrict__ Whp,
                                     const float* __restrict__ bh,
                                     float* __restrict__ ph)
{
  int wid  = (int)((blockIdx.x * (size_t)blockDim.x + threadIdx.x) >> 5);
  int lane = threadIdx.x & 31;
  if (wid >= 1024 * 8) return;          // 1024 m-tiles x 8 n-tiles
  int mt = wid >> 3, nt = wid & 7;
  v8f acc = {};
  for (int kb = 0; kb < FEATP_ / 32; ++kb) {
    v16bf a = ldA(feat, FEATP_, mt * 16, kb * 32);
    v16bf b = ldB(Whp, FEATP_, nt * 16, kb * 32);
    WMMA(acc, a, b);
  }
  int g = lane >> 4;
  int n = nt * 16 + (lane & 15);
  float bias = (n < 121) ? bh[n] : 0.0f;
#pragma unroll
  for (int j = 0; j < 8; ++j) {
    int m = mt * 16 + j + 8 * g;
    ph[(size_t)m * NHEAD_ + n] = acc[j] + bias;
  }
}

// ---------------- head post-processing into d_out ----------------

extern "C" __global__ void postproc(const float* __restrict__ ph,
                                    const float* __restrict__ phit,
                                    float* __restrict__ out)
{
  const size_t BT = (size_t)B_ * T_;
  float* means = out;                       // B,T,M,2
  float* sd    = means + BT * M_ * 2;       // B,T,M
  float* lw    = sd + BT * M_;              // B,T,M
  float* corr  = lw + BT * M_;              // B,T,M
  float* ll    = corr + BT * M_;            // B,T,1
  float* pt    = ll + BT;                   // B,1,65
  size_t i = (size_t)blockIdx.x * blockDim.x + threadIdx.x;
  if (i < BT) {
    const float* p = &ph[i * NHEAD_];       // i = b*T + t
    float mx = -1e30f;
    for (int m = 0; m < M_; ++m) mx = fmaxf(mx, p[1 + m]);
    float s = 0.0f;
    for (int m = 0; m < M_; ++m) s += __expf(p[1 + m] - mx);
    float lse = mx + __logf(s);
    for (int m = 0; m < M_; ++m) {
      lw[i * M_ + m]             = p[1 + m] - lse;
      means[(i * M_ + m) * 2]    = p[21 + 2 * m];
      means[(i * M_ + m) * 2 + 1] = p[21 + 2 * m + 1];
      sd[i * M_ + m]             = __expf(p[61 + m]);
      corr[i * M_ + m]           = tanhf(p[81 + m]);
    }
    ll[i] = p[0];
  }
  if (i < (size_t)B_ * 65) pt[i] = phit[i];
}

// ---------------- host launcher ----------------

extern "C" void kernel_launch(void* const* d_in, const int* in_sizes, int n_in,
                              void* d_out, int out_size, void* d_ws, size_t ws_size,
                              hipStream_t stream) {
  (void)in_sizes; (void)n_in; (void)out_size; (void)ws_size;
  const float* x    = (const float*)d_in[0];
  const int*   c    = (const int*)d_in[1];
  const float* Wih0 = (const float*)d_in[2];
  const float* Whh0 = (const float*)d_in[3];
  const float* bih0 = (const float*)d_in[4];
  const float* bhh0 = (const float*)d_in[5];
  const float* Wih1 = (const float*)d_in[6];
  const float* Whh1 = (const float*)d_in[7];
  const float* bih1 = (const float*)d_in[8];
  const float* bhh1 = (const float*)d_in[9];
  const float* Wih2 = (const float*)d_in[10];
  const float* Whh2 = (const float*)d_in[11];
  const float* bih2 = (const float*)d_in[12];
  const float* bhh2 = (const float*)d_in[13];
  const float* Ww   = (const float*)d_in[14];
  const float* bw   = (const float*)d_in[15];
  const float* Wh   = (const float*)d_in[16];
  const float* bh   = (const float*)d_in[17];

  char* ws = (char*)d_ws;
  size_t off = 0;
  auto take = [&](size_t bytes) -> void* {
    void* p = ws + off;
    off += (bytes + 255) & ~(size_t)255;
    return p;
  };
  const size_t BT = (size_t)B_ * T_;
  float*  out0  = (float*)take(BT * H_ * 4);
  float*  out1  = (float*)take(BT * H_ * 4);
  float*  out2  = (float*)take(BT * H_ * 4);
  float*  wseq  = (float*)take(BT * V_ * 4);
  float*  phit  = (float*)take((size_t)B_ * 65 * 4);
  float*  ph    = (float*)take(BT * NHEAD_ * 4);
  __bf16* Wih0p = (__bf16*)take((size_t)G3_ * IN0P_ * 2);
  __bf16* Whh0p = (__bf16*)take((size_t)G3_ * HP_ * 2);
  __bf16* Whh1p = (__bf16*)take((size_t)G3_ * HP_ * 2);
  __bf16* Whh2p = (__bf16*)take((size_t)G3_ * HP_ * 2);
  __bf16* Wih1p = (__bf16*)take((size_t)G3_ * INLP_ * 2);
  __bf16* Wih2p = (__bf16*)take((size_t)G3_ * INLP_ * 2);
  __bf16* Whp   = (__bf16*)take((size_t)NHEAD_ * FEATP_ * 2);
  __bf16* Ain   = (__bf16*)take(BT * INLP_ * 2);
  __bf16* feat  = (__bf16*)take(BT * FEATP_ * 2);

  auto pk = [&](__bf16* dst, const float* src, int rs, int cs, int rd, int cd) {
    int tot = rd * cd;
    pack_weight_bf16<<<(tot + 255) / 256, 256, 0, stream>>>(dst, src, rs, cs, rd, cd);
  };
  pk(Wih0p, Wih0, G3_, 63,  G3_, IN0P_);
  pk(Whh0p, Whh0, G3_, H_,  G3_, HP_);
  pk(Whh1p, Whh1, G3_, H_,  G3_, HP_);
  pk(Whh2p, Whh2, G3_, H_,  G3_, HP_);
  pk(Wih1p, Wih1, G3_, 463, G3_, INLP_);
  pk(Wih2p, Wih2, G3_, 463, G3_, INLP_);
  pk(Whp,   Wh,   121, G3_, NHEAD_, FEATP_);

  layer0_seq<<<1, 1024, 0, stream>>>(x, c, Wih0p, Whh0p, bih0, bhh0, Ww, bw,
                                     out0, wseq, phit);

  {
    size_t tot = BT * INLP_;
    pack_ain<<<(int)((tot + 255) / 256), 256, 0, stream>>>(Ain, x, out0, wseq);
  }
  layer_seq<<<1, 1024, 0, stream>>>(Ain, Wih1p, Whh1p, bih1, bhh1, out1);
  {
    size_t tot = BT * INLP_;
    pack_ain<<<(int)((tot + 255) / 256), 256, 0, stream>>>(Ain, x, out1, wseq);
  }
  layer_seq<<<1, 1024, 0, stream>>>(Ain, Wih2p, Whh2p, bih2, bhh2, out2);

  {
    size_t tot = BT * FEATP_;
    pack_feat<<<(int)((tot + 255) / 256), 256, 0, stream>>>(feat, out0, out1, out2);
  }
  head_gemm<<<2048, 128, 0, stream>>>(feat, Whp, bh, ph);
  postproc<<<(int)((BT + 255) / 256), 256, 0, stream>>>(ph, phit, (float*)d_out);
}
